// cpssloss_50672024159016
// MI455X (gfx1250) — matrix operand order repaired and checked by reading, not compiled, not run on hardware
//
#include <hip/hip_runtime.h>
#include <hip/hip_bf16.h>

typedef __attribute__((ext_vector_type(16))) _Float16 v16h;
typedef __attribute__((ext_vector_type(8)))  float    v8f;
typedef __attribute__((ext_vector_type(4)))  float    f4;

#define NP    1024        // P = (160/5)^2
#define NCLS  19
#define NBLK  50          // b * bs * bs = 2*5*5
#define RT    16          // row tiles per block (64 rows each)
#define NCH   32          // K-chunks of 32 columns

#define LOG2E 1.4426950408889634f
#define LN2   0.6931471805599453f

// Raw hardware transcendentals (base-2): v_exp_f32 / v_log_f32
#define EXP2(x) __builtin_amdgcn_exp2f(x)
#define LOG2(x) __builtin_amdgcn_logf(x)

__global__ void cpss_zero(float* out) {
    if (threadIdx.x < 2) out[threadIdx.x] = 0.0f;
}

__global__ __launch_bounds__(128) void cpss_main(
    const float* __restrict__ fuzzy,   // [2,5,5,1024,1024] f32
    const int*   __restrict__ label,   // [2,160,160] i32
    float*       __restrict__ out)     // [2] : {bce, ce}
{
    __shared__ int   s_gtf[NP];
    __shared__ int   s_cnt[32];
    __shared__ v16h  s_b0[NCH][32];    // one-hot B tiles, classes 0..15  (32 KB)
    __shared__ v16h  s_b1[NCH][32];    // one-hot B tiles, classes 16..31 (32 KB)
    __shared__ float s_num[64][33];    // 64 rows x 32 class slots (+pad)
    __shared__ float s_rsq[64];
    __shared__ float s_rbce[64];
    __shared__ float s_acc[2];

    const int tid  = threadIdx.x;
    const int wave = tid >> 5;
    const int lane = tid & 31;
    const int m    = lane & 15;        // row-in-tile (A) / class column (B)
    const int half = lane >> 4;        // K-half selector per WMMA layout

    const int rt  = blockIdx.x & (RT - 1);   // row tile 0..15 (64 rows)
    const int blk = blockIdx.x >> 4;         // 0..49 -> (b,s,t)
    const int t   = blk % 5;
    const int s   = (blk / 5) % 5;
    const int bb  = blk / 25;

    // ---- block labels gtf[q] and per-class counts into LDS ----
    for (int q = tid; q < NP; q += 128)
        s_gtf[q] = label[(bb * 160 + s * 32 + (q >> 5)) * 160 + t * 32 + (q & 31)];
    if (tid < 32) s_cnt[tid] = 0;
    if (tid < 2)  s_acc[tid] = 0.0f;
    __syncthreads();
    for (int q = tid; q < NP; q += 128) atomicAdd(&s_cnt[s_gtf[q]], 1);

    // ---- precompute one-hot B tiles once per WG (B depends only on kb,lane) ----
    for (int ci = wave; ci < NCH; ci += 4) {
        const int kb = ci * 32;
        v16h b0, b1;
#pragma unroll
        for (int i = 0; i < 16; ++i) {
            // B = onehot(gtf), 32x16 f16 layout: elem i -> K 8*half+i (i<8), 16+8*half+(i-8) (i>=8)
            const int gq = s_gtf[kb + half * 8 + i + ((i >= 8) ? 8 : 0)];
            b0[i] = (_Float16)((gq == m)      ? 1.0f : 0.0f);
            b1[i] = (_Float16)((gq == m + 16) ? 1.0f : 0.0f);
        }
        s_b0[ci][lane] = b0;
        s_b1[ci][lane] = b1;
    }
    __syncthreads();

    // ---- fused pass: BCE + row sum(sig^2) + WMMA (sig x onehot) ----
    const int rowloc = wave * 16 + m;        // 0..63 within WG
    const int prow   = rt * 64 + rowloc;     // row p within block
    const int myg    = s_gtf[prow];
    const float* rowptr = fuzzy + ((size_t)blk * NP + prow) * (size_t)NP;

    v8f c0 = {};  // num[p, 0..15]
    v8f c1 = {};  // num[p, 16..31] (19..31 identically zero)
    float bce = 0.0f, sq = 0.0f;

    for (int ci = 0; ci < NCH; ++ci) {
        const int kb = ci * 32;
        const int q0 = kb + half * 8;        // A-layout K groups per lane
        f4 xa = *(const f4*)(rowptr + q0);
        f4 xb = *(const f4*)(rowptr + q0 + 4);
        f4 xc = *(const f4*)(rowptr + q0 + 16);
        f4 xd = *(const f4*)(rowptr + q0 + 20);
        float xv[16] = {xa.x, xa.y, xa.z, xa.w,  xb.x, xb.y, xb.z, xb.w,
                        xc.x, xc.y, xc.z, xc.w,  xd.x, xd.y, xd.z, xd.w};
        v16h a;
#pragma unroll
        for (int i = 0; i < 16; ++i) {
            const float x  = xv[i];
            // exp(-|x|) via raw v_exp_f32; log1p via raw v_log_f32 (base-2 form)
            const float e  = EXP2(-fabsf(x) * LOG2E);
            const float r  = __builtin_amdgcn_rcpf(1.0f + e);   // raw v_rcp_f32
            const float sg = (x >= 0.0f) ? r : e * r;           // sigmoid(x)
            const float l  = LN2 * LOG2(1.0f + e);              // log1p(exp(-|x|))
            const int   qa = q0 + i + ((i >= 8) ? 8 : 0);
            const float z  = (s_gtf[qa] == myg) ? 1.0f : 0.0f;
            bce += fmaxf(x, 0.0f) - x * z + l;
            sq  += sg * sg;
            a[i] = (_Float16)sg;
        }
        const v16h b0v = s_b0[ci][lane];
        const v16h b1v = s_b1[ci][lane];
        c0 = __builtin_amdgcn_wmma_f32_16x16x32_f16(false, a, false, b0v,
                                                    (short)0, c0, false, false);
        c1 = __builtin_amdgcn_wmma_f32_16x16x32_f16(false, a, false, b1v,
                                                    (short)0, c1, false, false);
    }

    // combine the two K-halves of each row's scalar stats
    bce += __shfl_xor(bce, 16, 32);
    sq  += __shfl_xor(sq, 16, 32);

    // C/D layout: lane L, VGPR r -> (M = r + 8*(L/16), N = L%16)
#pragma unroll
    for (int r = 0; r < 8; ++r) {
        const int rr = wave * 16 + half * 8 + r;
        s_num[rr][m]      = c0[r];
        s_num[rr][16 + m] = c1[r];
    }
    if (half == 0) { s_rsq[rowloc] = sq; s_rbce[rowloc] = bce; }
    __syncthreads();

    // ---- epilogue: cosine-normalize + 19-class CE (online logsumexp, base 2) ----
    if (tid < 64) {
        const int pr = rt * 64 + tid;
        const int g  = s_gtf[pr];
        const float n1 = __builtin_amdgcn_sqrtf(s_rsq[tid]);
        float mx = -1e30f, sum = 0.0f, tv = 0.0f;
#pragma unroll
        for (int c = 0; c < NCLS; ++c) {
            const float n2 = __builtin_amdgcn_sqrtf((float)s_cnt[c]);
            const float d  = fmaxf(n1 * n2, 1e-8f);
            const float v  = s_num[tid][c] * __builtin_amdgcn_rcpf(d);
            tv = (c == g) ? v : tv;
            if (v > mx) { sum = sum * EXP2((mx - v) * LOG2E) + 1.0f; mx = v; }
            else        { sum += EXP2((v - mx) * LOG2E); }
        }
        const float ce = (mx + LN2 * LOG2(sum)) - tv;
        atomicAdd(&s_acc[0], s_rbce[tid]);
        atomicAdd(&s_acc[1], ce);
    }
    __syncthreads();
    if (tid == 0) {
        atomicAdd(&out[0], s_acc[0] * (1.0f / 52428800.0f)); // / (2*25*1024*1024)
        atomicAdd(&out[1], s_acc[1] * (1.0f / 51200.0f));    // / (2*160*160)
    }
}

extern "C" void kernel_launch(void* const* d_in, const int* in_sizes, int n_in,
                              void* d_out, int out_size, void* d_ws, size_t ws_size,
                              hipStream_t stream) {
    (void)in_sizes; (void)n_in; (void)out_size; (void)d_ws; (void)ws_size;
    const float* fuzzy = (const float*)d_in[0];
    const int*   label = (const int*)d_in[1];
    float* out = (float*)d_out;
    cpss_zero<<<1, 32, 0, stream>>>(out);
    cpss_main<<<NBLK * RT, 128, 0, stream>>>(fuzzy, label, out);
}
